// LSTMSeq2Seq_14585708937644
// MI455X (gfx1250) — matrix operand order repaired and checked by reading, not compiled, not run on hardware
//
#include <hip/hip_runtime.h>
#include <hip/hip_bf16.h>

// ---------------------------------------------------------------------------
// LSTM seq2seq w/ attention on gfx1250 (MI455X).
//   - LSTM step = ONE fused kernel per layer: each wave computes the same
//     16x16 (batch, j) patch for all four gate blocks (i/f/g/o) with
//     v_wmma_f32_16x16x32_bf16 (A frag reused 4x), then applies the gate
//     nonlinearities and updates c (f32) / h (bf16) in-register.
//   - h is ping-pong double-buffered (fused kernel reads old h as WMMA A
//     operand while writing new h); c updates in place (1 lane per element).
//   - Projections (q, Wo, fc1, K, V) use a generic WMMA GEMM.
//   - Attention (per-batch GEMV + softmax) done VALU with wave32 shfl_xor.
// ---------------------------------------------------------------------------

typedef __attribute__((ext_vector_type(16))) __bf16 v16bf;
typedef __attribute__((ext_vector_type(8)))  __bf16 v8bf;
typedef __attribute__((ext_vector_type(8)))  float  v8f;

static constexpr int kB   = 128;   // batch
static constexpr int kT   = 256;   // encoder sequence length
static constexpr int kH   = 512;   // hidden
static constexpr int kG   = 2048;  // 4*H gate width
static constexpr int kNH  = 4;     // heads
static constexpr int kHD  = 128;   // head dim
static constexpr int kHor = 30;    // decode horizon
static constexpr int kFC1 = 256;   // fc hidden

// ---------------------------------------------------------------------------
// WMMA fragment loaders (contiguous k order, same for A and B, wave32).
// A (16x32 bf16): lane row = m0+(lane&15); 16 contiguous K
//                 at offset 0 / 16 by lane group.
// B (32x16 from row-major W[N,K]): lane row = n0+(lane&15); 16 contiguous K
//                 at offset 0 / 16 by lane group.
// ---------------------------------------------------------------------------
__device__ __forceinline__ v16bf ld_afrag(const __bf16* p)
{
    v16bf a;
    *(v8bf*)&a       = *(const v8bf*)(p);
    *((v8bf*)&a + 1) = *(const v8bf*)(p + 8);
    return a;
}
__device__ __forceinline__ v16bf ld_bfrag(const __bf16* p)
{
    v16bf b;
    *(v8bf*)&b       = *(const v8bf*)(p);
    *((v8bf*)&b + 1) = *(const v8bf*)(p + 8);
    return b;
}
__device__ __forceinline__ float sigm(float x) { return 1.0f / (1.0f + __expf(-x)); }

// ---------------------------------------------------------------------------
// Fused LSTM step kernel (one layer):
//   gates(b, g*512+j) = A1 @ W1^T (+ A2 @ W2^T), g = i,f,g,o
//   c = sig(f)*c + sig(i)*tanh(g');  hNew = sig(o)*tanh(c)
// Each wave: one 16x16 (b,j) patch, 4 gate accumulators sharing the A frag.
// din/dinW supply the decoder layer-0 K=1 input term. hNew must not alias
// A1/A2 (double-buffered by the host).
// ---------------------------------------------------------------------------
__global__ __launch_bounds__(256)
void k_lstm_fused(const __bf16* __restrict__ A1, int lda1,
                  const __bf16* __restrict__ W1, int K1,
                  const __bf16* __restrict__ A2, int lda2,
                  const __bf16* __restrict__ W2, int K2,
                  const float* __restrict__ bias,
                  const float* __restrict__ din, const float* __restrict__ dinW,
                  float* __restrict__ c, __bf16* __restrict__ hNew,
                  __bf16* __restrict__ encout, int encStride)
{
    const int lane   = threadIdx.x & 31;
    const int wave   = blockIdx.x * (blockDim.x >> 5) + (threadIdx.x >> 5);
    const int nwaves = gridDim.x * (blockDim.x >> 5);
    const int ntn    = kH >> 4;                    // 32 j-tiles
    const int ntiles = (kB >> 4) * ntn;            // 256 tiles
    const int rsel   = lane >> 4;
    const int lrow   = lane & 15;

    for (int tile = wave; tile < ntiles; tile += nwaves) {
        const int tm = tile / ntn;
        const int tn = tile - tm * ntn;
        const int m0 = tm << 4, n0 = tn << 4;
        v8f acc0 = {}, acc1 = {}, acc2 = {}, acc3 = {};

        {
            const __bf16* Ab  = A1 + (size_t)(m0 + lrow) * lda1 + (rsel << 4);
            const __bf16* Wb0 = W1 + (size_t)(0 * kH + n0 + lrow) * K1 + (rsel << 4);
            const __bf16* Wb1 = W1 + (size_t)(1 * kH + n0 + lrow) * K1 + (rsel << 4);
            const __bf16* Wb2 = W1 + (size_t)(2 * kH + n0 + lrow) * K1 + (rsel << 4);
            const __bf16* Wb3 = W1 + (size_t)(3 * kH + n0 + lrow) * K1 + (rsel << 4);
            for (int k = 0; k < K1; k += 32) {
                __builtin_prefetch(Wb0 + k + 256, 0, 1);   // global_prefetch_b8
                v16bf a = ld_afrag(Ab + k);
                acc0 = __builtin_amdgcn_wmma_f32_16x16x32_bf16(
                           false, a, false, ld_bfrag(Wb0 + k), (short)0, acc0, false, false);
                acc1 = __builtin_amdgcn_wmma_f32_16x16x32_bf16(
                           false, a, false, ld_bfrag(Wb1 + k), (short)0, acc1, false, false);
                acc2 = __builtin_amdgcn_wmma_f32_16x16x32_bf16(
                           false, a, false, ld_bfrag(Wb2 + k), (short)0, acc2, false, false);
                acc3 = __builtin_amdgcn_wmma_f32_16x16x32_bf16(
                           false, a, false, ld_bfrag(Wb3 + k), (short)0, acc3, false, false);
            }
        }
        if (A2) {
            const __bf16* Ab  = A2 + (size_t)(m0 + lrow) * lda2 + (rsel << 4);
            const __bf16* Wb0 = W2 + (size_t)(0 * kH + n0 + lrow) * K2 + (rsel << 4);
            const __bf16* Wb1 = W2 + (size_t)(1 * kH + n0 + lrow) * K2 + (rsel << 4);
            const __bf16* Wb2 = W2 + (size_t)(2 * kH + n0 + lrow) * K2 + (rsel << 4);
            const __bf16* Wb3 = W2 + (size_t)(3 * kH + n0 + lrow) * K2 + (rsel << 4);
            for (int k = 0; k < K2; k += 32) {
                __builtin_prefetch(Wb0 + k + 256, 0, 1);
                v16bf a = ld_afrag(Ab + k);
                acc0 = __builtin_amdgcn_wmma_f32_16x16x32_bf16(
                           false, a, false, ld_bfrag(Wb0 + k), (short)0, acc0, false, false);
                acc1 = __builtin_amdgcn_wmma_f32_16x16x32_bf16(
                           false, a, false, ld_bfrag(Wb1 + k), (short)0, acc1, false, false);
                acc2 = __builtin_amdgcn_wmma_f32_16x16x32_bf16(
                           false, a, false, ld_bfrag(Wb2 + k), (short)0, acc2, false, false);
                acc3 = __builtin_amdgcn_wmma_f32_16x16x32_bf16(
                           false, a, false, ld_bfrag(Wb3 + k), (short)0, acc3, false, false);
            }
        }

        // ---- fused gate elementwise ---------------------------------------
        // C/D layout: element r -> (row = m0 + r + 8*rsel, col = n0 + (lane&15))
        const int col   = n0 + lrow;
        const int rbase = m0 + (rsel << 3);
        const float bi = bias[col];
        const float bf = bias[kH + col];
        const float bg = bias[2 * kH + col];
        const float bo = bias[3 * kH + col];
        float wi = 0.f, wf = 0.f, wg = 0.f, wo = 0.f;
        if (dinW) {
            wi = dinW[col];          wf = dinW[kH + col];
            wg = dinW[2 * kH + col]; wo = dinW[3 * kH + col];
        }
        #pragma unroll
        for (int r = 0; r < 8; ++r) {
            const int row = rbase + r;
            float gi = acc0[r] + bi;
            float gf = acc1[r] + bf;
            float gg = acc2[r] + bg;
            float go = acc3[r] + bo;
            if (dinW) {
                const float dv = din[row];
                gi += dv * wi; gf += dv * wf; gg += dv * wg; go += dv * wo;
            }
            const size_t off = (size_t)row * kH + col;
            const float cn = sigm(gf) * c[off] + sigm(gi) * tanhf(gg);
            c[off] = cn;
            const float hn = sigm(go) * tanhf(cn);
            hNew[off] = (__bf16)hn;
            if (encout) encout[(size_t)row * encStride + col] = (__bf16)hn;
        }
    }
}

// ---------------------------------------------------------------------------
// Generic WMMA GEMM:  C[M,N] = A[M,K] @ W[N,K]^T + bias (relu?) -> f32/bf16
// ---------------------------------------------------------------------------
__global__ __launch_bounds__(256)
void k_gemm_wmma(const __bf16* __restrict__ A, int lda,
                 const __bf16* __restrict__ W, int K,
                 const float* __restrict__ bias,
                 int M, int N, int relu,
                 float* __restrict__ Cf, __bf16* __restrict__ Cb)
{
    const int lane   = threadIdx.x & 31;
    const int wave   = blockIdx.x * (blockDim.x >> 5) + (threadIdx.x >> 5);
    const int nwaves = gridDim.x * (blockDim.x >> 5);
    const int ntn    = N >> 4;
    const int ntiles = (M >> 4) * ntn;
    const int rsel   = lane >> 4;
    const int lrow   = lane & 15;

    for (int tile = wave; tile < ntiles; tile += nwaves) {
        const int tm = tile / ntn;
        const int tn = tile - tm * ntn;
        const int m0 = tm << 4, n0 = tn << 4;
        v8f acc = {};
        const __bf16* Ab = A + (size_t)(m0 + lrow) * lda + (rsel << 4);
        const __bf16* Wb = W + (size_t)(n0 + lrow) * K + (rsel << 4);
        for (int k = 0; k < K; k += 32) {
            __builtin_prefetch(Wb + k + 256, 0, 1);
            acc = __builtin_amdgcn_wmma_f32_16x16x32_bf16(
                      false, ld_afrag(Ab + k), false, ld_bfrag(Wb + k),
                      (short)0, acc, false, false);
        }
        const int   col   = n0 + lrow;
        const float bv    = bias ? bias[col] : 0.0f;
        const int   rbase = m0 + (rsel << 3);
        #pragma unroll
        for (int r = 0; r < 8; ++r) {
            float v = acc[r] + bv;
            if (relu) v = v > 0.0f ? v : 0.0f;
            const size_t off = (size_t)(rbase + r) * N + col;
            if (Cf) Cf[off] = v;
            if (Cb) Cb[off] = (__bf16)v;
        }
    }
}

// ---------------------------------------------------------------------------
// Attention: one block per batch element, one wave per head (wave32).
// ---------------------------------------------------------------------------
__global__ __launch_bounds__(128)
void k_attention(const float* __restrict__ qf,
                 const __bf16* __restrict__ Kb, const __bf16* __restrict__ Vb,
                 __bf16* __restrict__ aob,
                 float* __restrict__ awBase, int awStride)
{
    const int b    = blockIdx.x;
    const int h    = threadIdx.x >> 5;
    const int lane = threadIdx.x & 31;
    __shared__ float wsm[kNH][kT];
    __shared__ float qsm[kNH][kHD];

    for (int i = lane; i < kHD; i += 32)
        qsm[h][i] = qf[(size_t)b * kH + h * kHD + i];
    __syncthreads();

    const float scale = 0.08838834764831845f;  // 1/sqrt(128)
    float s[8];
    float mx = -1e30f;
    #pragma unroll
    for (int i = 0; i < 8; ++i) {
        const int l = lane * 8 + i;
        const __bf16* kp = Kb + ((size_t)b * kT + l) * kH + h * kHD;
        float acc = 0.0f;
        for (int d = 0; d < kHD; d += 8) {
            v8bf kv = *(const v8bf*)(kp + d);
            #pragma unroll
            for (int e = 0; e < 8; ++e) acc += qsm[h][d + e] * (float)kv[e];
        }
        s[i] = acc * scale;
        mx = fmaxf(mx, s[i]);
    }
    for (int off = 16; off; off >>= 1) mx = fmaxf(mx, __shfl_xor(mx, off, 32));
    float sum = 0.0f;
    #pragma unroll
    for (int i = 0; i < 8; ++i) { s[i] = __expf(s[i] - mx); sum += s[i]; }
    for (int off = 16; off; off >>= 1) sum += __shfl_xor(sum, off, 32);
    const float inv = 1.0f / sum;
    #pragma unroll
    for (int i = 0; i < 8; ++i) wsm[h][lane * 8 + i] = s[i] * inv;
    __syncthreads();

    float a0 = 0, a1 = 0, a2 = 0, a3 = 0;
    for (int l = 0; l < kT; ++l) {
        const float wl = wsm[h][l];
        const __bf16* vp = Vb + ((size_t)b * kT + l) * kH + h * kHD;
        a0 += wl * (float)vp[lane];
        a1 += wl * (float)vp[lane + 32];
        a2 += wl * (float)vp[lane + 64];
        a3 += wl * (float)vp[lane + 96];
    }
    __bf16* ap = aob + (size_t)b * kH + h * kHD;
    ap[lane]      = (__bf16)a0;
    ap[lane + 32] = (__bf16)a1;
    ap[lane + 64] = (__bf16)a2;
    ap[lane + 96] = (__bf16)a3;

    for (int l = threadIdx.x; l < kT; l += 128)
        awBase[(size_t)b * awStride + l] =
            0.25f * (wsm[0][l] + wsm[1][l] + wsm[2][l] + wsm[3][l]);
}

// ---------------------------------------------------------------------------
// Final fc2 (N=3, too narrow for WMMA): pred -> d_out, and pred[:,1] -> din.
// ---------------------------------------------------------------------------
__global__ __launch_bounds__(128)
void k_fc2(const __bf16* __restrict__ fc1b, const float* __restrict__ W2,
           const float* __restrict__ b2, float* __restrict__ outBase,
           int outStride, float* __restrict__ din)
{
    const int idx = blockIdx.x * blockDim.x + threadIdx.x;
    if (idx >= kB * 3) return;
    const int b = idx / 3, j = idx - b * 3;
    const __bf16* f = fc1b + (size_t)b * kFC1;
    const float*  w = W2 + (size_t)j * kFC1;
    float acc = b2[j];
    for (int k = 0; k < kFC1; k += 8) {
        v8bf fv = *(const v8bf*)(f + k);
        #pragma unroll
        for (int e = 0; e < 8; ++e) acc += (float)fv[e] * w[k + e];
    }
    outBase[(size_t)b * outStride + j] = acc;
    if (j == 1) din[b] = acc;
}

// f32 -> bf16 convert
__global__ __launch_bounds__(256)
void k_cvt(const float* __restrict__ s, __bf16* __restrict__ d, int n)
{
    const int i = blockIdx.x * blockDim.x + threadIdx.x;
    if (i < n) d[i] = (__bf16)s[i];
}

// zero recurrent state (both ping-pong copies) + load din from last_target
__global__ __launch_bounds__(256)
void k_init(float* __restrict__ c0, float* __restrict__ c1,
            __bf16* __restrict__ h0a, __bf16* __restrict__ h0b,
            __bf16* __restrict__ h1a, __bf16* __restrict__ h1b,
            const float* __restrict__ last_target, float* __restrict__ din)
{
    const int idx = blockIdx.x * blockDim.x + threadIdx.x;
    if (idx < kB * kH) {
        c0[idx] = 0.0f; c1[idx] = 0.0f;
        h0a[idx] = (__bf16)0.0f; h0b[idx] = (__bf16)0.0f;
        h1a[idx] = (__bf16)0.0f; h1b[idx] = (__bf16)0.0f;
    }
    if (idx < kB) din[idx] = last_target[idx];
}

// ---------------------------------------------------------------------------
// Host side
// ---------------------------------------------------------------------------
static inline void launch_gemm(hipStream_t s,
                               const __bf16* A, int lda, const __bf16* W, int K,
                               const float* bias, int M, int N, int relu,
                               float* Cf, __bf16* Cb)
{
    const int tiles = (M / 16) * (N / 16);
    int blocks = (tiles + 7) / 8;
    if (blocks > 1024) blocks = 1024;
    hipLaunchKernelGGL(k_gemm_wmma, dim3(blocks), dim3(256), 0, s,
                       A, lda, W, K, bias, M, N, relu, Cf, Cb);
}

static inline void launch_lstm(hipStream_t s,
                               const __bf16* A1, int lda1, const __bf16* W1, int K1,
                               const __bf16* A2, int lda2, const __bf16* W2, int K2,
                               const float* bias, const float* din, const float* dinW,
                               float* c, __bf16* hNew, __bf16* encout, int encStride)
{
    hipLaunchKernelGGL(k_lstm_fused, dim3(32), dim3(256), 0, s,   // 256 tiles / 8 waves
                       A1, lda1, W1, K1, A2, lda2, W2, K2,
                       bias, din, dinW, c, hNew, encout, encStride);
}

static inline void launch_cvt(hipStream_t s, const float* src, __bf16* dst, int n)
{
    hipLaunchKernelGGL(k_cvt, dim3((n + 255) / 256), dim3(256), 0, s, src, dst, n);
}

extern "C" void kernel_launch(void* const* d_in, const int* in_sizes, int n_in,
                              void* d_out, int out_size, void* d_ws, size_t ws_size,
                              hipStream_t stream)
{
    const float* x        = (const float*)d_in[0];
    const float* last_tgt = (const float*)d_in[1];
    const float* eWih0    = (const float*)d_in[2];
    const float* eWhh0    = (const float*)d_in[3];
    const float* eb0      = (const float*)d_in[4];
    const float* eWih1    = (const float*)d_in[5];
    const float* eWhh1    = (const float*)d_in[6];
    const float* eb1      = (const float*)d_in[7];
    const float* dWih0    = (const float*)d_in[8];
    const float* dWhh0    = (const float*)d_in[9];
    const float* db0      = (const float*)d_in[10];
    const float* dWih1    = (const float*)d_in[11];
    const float* dWhh1    = (const float*)d_in[12];
    const float* db1      = (const float*)d_in[13];
    const float* Wq       = (const float*)d_in[14];
    const float* Wk       = (const float*)d_in[15];
    const float* Wv       = (const float*)d_in[16];
    const float* Wo       = (const float*)d_in[17];
    const float* bq       = (const float*)d_in[18];
    const float* bk       = (const float*)d_in[19];
    const float* bv       = (const float*)d_in[20];
    const float* bo       = (const float*)d_in[21];
    const float* fcW1     = (const float*)d_in[22];
    const float* fcb1     = (const float*)d_in[23];
    const float* fcW2     = (const float*)d_in[24];
    const float* fcb2     = (const float*)d_in[25];

    float* outPred = (float*)d_out;                        // (128,30,3)
    float* outAttn = outPred + (size_t)kB * kHor * 3;      // (128,30,256)

    char* wp = (char*)d_ws;
    auto carve = [&](size_t bytes) -> void* {
        void* r = (void*)wp;
        wp += (bytes + 255) & ~(size_t)255;
        return r;
    };
    __bf16* xb      = (__bf16*)carve((size_t)kB * kT * 32 * 2);
    __bf16* eWih0b  = (__bf16*)carve((size_t)kG * 32 * 2);
    __bf16* eWhh0b  = (__bf16*)carve((size_t)kG * kH * 2);
    __bf16* eWih1b  = (__bf16*)carve((size_t)kG * kH * 2);
    __bf16* eWhh1b  = (__bf16*)carve((size_t)kG * kH * 2);
    __bf16* dWhh0b  = (__bf16*)carve((size_t)kG * kH * 2);
    __bf16* dWih1b  = (__bf16*)carve((size_t)kG * kH * 2);
    __bf16* dWhh1b  = (__bf16*)carve((size_t)kG * kH * 2);
    __bf16* Wqb     = (__bf16*)carve((size_t)kH * kH * 2);
    __bf16* Wkb     = (__bf16*)carve((size_t)kH * kH * 2);
    __bf16* Wvb     = (__bf16*)carve((size_t)kH * kH * 2);
    __bf16* Wob     = (__bf16*)carve((size_t)kH * kH * 2);
    __bf16* fcW1b   = (__bf16*)carve((size_t)kFC1 * kH * 2);
    __bf16* h0p[2]  = { (__bf16*)carve((size_t)kB * kH * 2),
                        (__bf16*)carve((size_t)kB * kH * 2) };
    __bf16* h1p[2]  = { (__bf16*)carve((size_t)kB * kH * 2),
                        (__bf16*)carve((size_t)kB * kH * 2) };
    __bf16* encout  = (__bf16*)carve((size_t)kB * kT * kH * 2);
    __bf16* Kbuf    = (__bf16*)carve((size_t)kB * kT * kH * 2);
    __bf16* Vbuf    = (__bf16*)carve((size_t)kB * kT * kH * 2);
    __bf16* aob     = (__bf16*)carve((size_t)kB * kH * 2);
    __bf16* aopb    = (__bf16*)carve((size_t)kB * kH * 2);
    __bf16* fc1b    = (__bf16*)carve((size_t)kB * kFC1 * 2);
    float*  c0      = (float*)carve((size_t)kB * kH * 4);
    float*  c1      = (float*)carve((size_t)kB * kH * 4);
    float*  qf      = (float*)carve((size_t)kB * kH * 4);
    float*  din     = (float*)carve((size_t)kB * 4);

    // one-time converts (f32 -> bf16)
    launch_cvt(stream, x,     xb,     kB * kT * 32);
    launch_cvt(stream, eWih0, eWih0b, kG * 32);
    launch_cvt(stream, eWhh0, eWhh0b, kG * kH);
    launch_cvt(stream, eWih1, eWih1b, kG * kH);
    launch_cvt(stream, eWhh1, eWhh1b, kG * kH);
    launch_cvt(stream, dWhh0, dWhh0b, kG * kH);
    launch_cvt(stream, dWih1, dWih1b, kG * kH);
    launch_cvt(stream, dWhh1, dWhh1b, kG * kH);
    launch_cvt(stream, Wq,    Wqb,    kH * kH);
    launch_cvt(stream, Wk,    Wkb,    kH * kH);
    launch_cvt(stream, Wv,    Wvb,    kH * kH);
    launch_cvt(stream, Wo,    Wob,    kH * kH);
    launch_cvt(stream, fcW1,  fcW1b,  kFC1 * kH);

    hipLaunchKernelGGL(k_init, dim3((kB * kH + 255) / 256), dim3(256), 0, stream,
                       c0, c1, h0p[0], h0p[1], h1p[0], h1p[1], last_tgt, din);

    // ---- encoder: 256 dependent steps, 2 fused kernels each ---------------
    for (int t = 0; t < kT; ++t) {
        const int cur = t & 1, nxt = cur ^ 1;
        // layer 0: gates = x_t @ Wih0^T + h0 @ Whh0^T
        launch_lstm(stream, xb + t * 32, kT * 32, eWih0b, 32,
                    h0p[cur], kH, eWhh0b, kH,
                    eb0, nullptr, nullptr, c0, h0p[nxt], nullptr, 0);
        // layer 1: gates = h0_new @ Wih1^T + h1 @ Whh1^T ; h1 -> enc_out[:,t,:]
        launch_lstm(stream, h0p[nxt], kH, eWih1b, kH,
                    h1p[cur], kH, eWhh1b, kH,
                    eb1, nullptr, nullptr, c1, h1p[nxt],
                    encout + (size_t)t * kH, kT * kH);
    }
    // after t = 255 (cur=1), final states live in index 0

    // ---- K / V projections over all (b,t) ---------------------------------
    launch_gemm(stream, encout, kH, Wkb, kH, bk, kB * kT, kH, 0, nullptr, Kbuf);
    launch_gemm(stream, encout, kH, Wvb, kH, bv, kB * kT, kH, 0, nullptr, Vbuf);

    // ---- decoder: 30 dependent steps --------------------------------------
    for (int st = 0; st < kHor; ++st) {
        const int cur = st & 1, nxt = cur ^ 1;
        // layer 0: gates = h0 @ Whh0^T ; din*Wih0 folded into elementwise
        launch_lstm(stream, h0p[cur], kH, dWhh0b, kH,
                    nullptr, 0, nullptr, 0,
                    db0, din, dWih0, c0, h0p[nxt], nullptr, 0);
        // layer 1
        launch_lstm(stream, h0p[nxt], kH, dWih1b, kH,
                    h1p[cur], kH, dWhh1b, kH,
                    db1, nullptr, nullptr, c1, h1p[nxt], nullptr, 0);
        // q = h1_new @ Wq^T + bq
        launch_gemm(stream, h1p[nxt], kH, Wqb, kH, bq, kB, kH, 0, qf, nullptr);
        // attention
        hipLaunchKernelGGL(k_attention, dim3(kB), dim3(128), 0, stream,
                           qf, Kbuf, Vbuf, aob,
                           outAttn + (size_t)st * kT, kHor * kT);
        // out proj + fc1(relu) + fc2
        launch_gemm(stream, aob,  kH, Wob,   kH, bo,   kB, kH,   0, nullptr, aopb);
        launch_gemm(stream, aopb, kH, fcW1b, kH, fcb1, kB, kFC1, 1, nullptr, fc1b);
        hipLaunchKernelGGL(k_fc2, dim3((kB * 3 + 127) / 128), dim3(128), 0, stream,
                           fc1b, fcW2, fcb2, outPred + (size_t)st * 3, kHor * 3, din);
    }
}
